// MLA_3367254360088
// MI455X (gfx1250) — compile-verified, matrix operand order checked by reading, not run on hardware
//
#include <hip/hip_runtime.h>
#include <hip/hip_bf16.h>

// ---------------------------------------------------------------------------
// MLA-style block: RMSNorm -> Q/KV proj (bf16 WMMA) -> RoPE -> 9x9 per-token
// head attention (f32 VALU) -> out proj (bf16 WMMA) + residual.
// Target: gfx1250 (wave32, v_wmma_f32_16x16x32_bf16).
// ---------------------------------------------------------------------------

typedef __attribute__((ext_vector_type(16))) __bf16 v16bf;
typedef __attribute__((ext_vector_type(8)))  float  v8f;

#define D_MODEL   576
#define KV_OUT    1152
#define N_HEADS   9
#define HEAD_DIM  64
#define SEQ_LEN   2048
#define M_BLOCK   32          // tokens per workgroup
#define KCH       18          // 576 / 32 K-chunks
#define FRAG_ELEM 512         // 16x32 bf16 fragment = 512 elems = 1KB

// packed-weight fragment table (all weights share K = 576):
//   Wq : ntiles 36  -> frags [0,    648)
//   Wkv: ntiles 72  -> frags [648, 1944)
//   Wo : ntiles 36  -> frags [1944,2592)
#define QF   (36 * KCH)
#define KVF  (72 * KCH)
#define OF   (36 * KCH)
#define NFRAG (QF + KVF + OF)          // 2592
#define PACK_ELEMS (NFRAG * FRAG_ELEM) // 1,327,104 bf16 (~2.6 MB)

// LDS layout (bytes)
#define XNA_OFF 0                       // A-fragments of xn (aliased by attn)
#define XNA_SZ  (2 * KCH * 1024)        // 36864
#define QS_OFF  (XNA_OFF + XNA_SZ)      // q rows,  32x576 bf16
#define KS_OFF  (QS_OFF + M_BLOCK * D_MODEL * 2)
#define VS_OFF  (KS_OFF + M_BLOCK * D_MODEL * 2)
#define RT_OFF  (VS_OFF + M_BLOCK * D_MODEL * 2)     // RoPE cos/sin table
#define RT_SZ   (M_BLOCK * (HEAD_DIM / 2) * 2 * 4)   // 32*32*float2 = 8192
#define LDS_BYTES (RT_OFF + RT_SZ)                   // 155648 (2 WGs/WGP)

// 16-bit A/B fragment element mapping (CDNA5 ISA):
//   kk in [0,32): grp = kk>>4 selects VGPR group (0..3 vs 4..7),
//   half = (kk&15)>>3 selects lane half, j = grp*4 + ((kk&7)>>1), p = kk&1.
__device__ __forceinline__ int frag_elem_off(int m_or_n, int kk) {
  int grp  = kk >> 4;
  int wi   = kk & 15;
  int half = wi >> 3;
  int j    = grp * 4 + ((wi & 7) >> 1);
  int p    = kk & 1;
  int fragLane = half * 16 + m_or_n;     // lane within wave
  return fragLane * 16 + j * 2 + p;      // element index inside 512-elem frag
}

// ---------------------------------------------------------------------------
// Prep: f32 weights -> bf16, pre-swizzled into B-fragment order.
// out[frag*512 + lane*16 + (j*2+p)] = W[n][k]
// ---------------------------------------------------------------------------
__global__ void pack_weights_kernel(const float* __restrict__ Wq,
                                    const float* __restrict__ Wkv,
                                    const float* __restrict__ Wo,
                                    __bf16* __restrict__ outp) {
  int idx = blockIdx.x * 256 + threadIdx.x;
  if (idx >= PACK_ELEMS) return;
  int frag = idx >> 9;
  int v    = idx & 511;

  const float* W;
  int fo = frag;
  if (fo < QF)            { W = Wq; }
  else if (fo < QF + KVF) { W = Wkv; fo -= QF; }
  else                    { W = Wo;  fo -= QF + KVF; }
  int nt = fo / KCH;
  int kc = fo % KCH;

  int lane = v >> 4;
  int e    = v & 15;
  int j    = e >> 1;
  int p    = e & 1;
  int n    = nt * 16 + (lane & 15);
  int half = lane >> 4;
  int kk   = (j >> 2) * 16 + half * 8 + (j & 3) * 2 + p;
  int k    = kc * 32 + kk;

  outp[idx] = (__bf16)W[(size_t)n * D_MODEL + k];
}

// ---------------------------------------------------------------------------
// Main fused kernel: 256 threads (8 waves), 32 tokens per block.
// ---------------------------------------------------------------------------
__global__ void __launch_bounds__(256)
mla_fused_kernel(const float* __restrict__ x,
                 const __bf16* __restrict__ pack,
                 float* __restrict__ out) {
  extern __shared__ __attribute__((aligned(32))) char smem[];
  __bf16* xnA = (__bf16*)(smem + XNA_OFF);  // A-fragments (later: attn frags)
  __bf16* qS  = (__bf16*)(smem + QS_OFF);
  __bf16* kS  = (__bf16*)(smem + KS_OFF);
  __bf16* vS  = (__bf16*)(smem + VS_OFF);
  float2* ropeT = (float2*)(smem + RT_OFF); // [token r][pair i] = {cos, sin}

  const int tid  = threadIdx.x;
  const int lane = tid & 31;
  const int wave = tid >> 5;
  const int blockTok = blockIdx.x * M_BLOCK;

  // ---------------- Stage A0: RoPE cos/sin table (1024 entries) -----------
  {
    #pragma unroll
    for (int t = 0; t < 4; ++t) {
      int idx = tid + 256 * t;                   // 0..1023
      int r = idx >> 5;                          // token in block
      int i = idx & 31;                          // rotary pair index
      int pos = (blockTok + r) & (SEQ_LEN - 1);  // blocks never straddle batch
      // theta_i = 10000^(-2i/64)
      float theta = __expf(-(float)(2 * i) * (1.0f / 64.0f) * 9.210340372f);
      float c, sn;
      sincosf((float)pos * theta, &sn, &c);
      ropeT[idx] = make_float2(c, sn);
    }
  }

  // ---------------- Stage A: RMSNorm -> packed bf16 A-fragments -----------
  {
    // element mapping is fixed per lane: kk == lane
    int grp = lane >> 4, wi = lane & 15, hf = wi >> 3;
    int jj  = grp * 4 + ((wi & 7) >> 1), pp = lane & 1;
    #pragma unroll
    for (int rr = 0; rr < 4; ++rr) {
      int row = wave + rr * 8;               // 0..31
      const float* xr = x + (size_t)(blockTok + row) * D_MODEL;
      float vals[KCH];
      float ss = 0.f;
      #pragma unroll
      for (int i = 0; i < KCH; ++i) {
        vals[i] = xr[lane + 32 * i];
        ss += vals[i] * vals[i];
      }
      #pragma unroll
      for (int off = 16; off > 0; off >>= 1) ss += __shfl_xor(ss, off, 32);
      float rn = rsqrtf(ss * (1.0f / (float)D_MODEL) + 1e-6f);

      int mtile = row >> 4, m = row & 15;
      int fragLane = hf * 16 + m;
      #pragma unroll
      for (int i = 0; i < KCH; ++i) {
        int frag = mtile * KCH + i;
        xnA[frag * FRAG_ELEM + fragLane * 16 + jj * 2 + pp] =
            (__bf16)(vals[i] * rn);
      }
    }
  }
  __syncthreads();

  // ---------------- Stage B: fused Q + KV projection (WMMA) ---------------
  // 108 n-tiles (36 q + 72 kv) x 2 m-tiles = 216 jobs; 27 per wave.
  {
    #pragma unroll 1
    for (int t = 0; t < 27; ++t) {
      int jb    = wave + 8 * t;
      int mtile = jb & 1;
      int nt    = jb >> 1;               // 0..107
      v8f acc = {};
      const __bf16* wbase =
          pack + (size_t)((nt < 36) ? nt * KCH : QF + (nt - 36) * KCH) * FRAG_ELEM;
      const __bf16* aA = xnA + mtile * KCH * FRAG_ELEM;
      #pragma unroll
      for (int k = 0; k < KCH; ++k) {
        v16bf a = *(const v16bf*)(aA    + k * FRAG_ELEM + lane * 16);
        v16bf b = *(const v16bf*)(wbase + k * FRAG_ELEM + lane * 16);
        acc = __builtin_amdgcn_wmma_f32_16x16x32_bf16(
            false, a, false, b, (short)0, acc, false, false);
      }
      // C layout: VGPR j, lanes0-15 -> M=j, lanes16-31 -> M=j+8; N = lane&15
      int colb = ((nt < 36) ? nt : nt - 36) * 16 + (lane & 15);
      int rowb = mtile * 16 + (lane >> 4) * 8;
      #pragma unroll
      for (int j = 0; j < 8; ++j) {
        int row = rowb + j;
        float val = acc[j];
        if (nt < 36) {
          qS[row * D_MODEL + colb] = (__bf16)val;
        } else {
          int g = colb >> 7, cc = colb & 127;   // kv = (head, 2*64)
          if (cc < HEAD_DIM) kS[row * D_MODEL + g * HEAD_DIM + cc] = (__bf16)val;
          else               vS[row * D_MODEL + g * HEAD_DIM + (cc - HEAD_DIM)] = (__bf16)val;
        }
      }
    }
  }
  __syncthreads();

  // ---------------- Stage C: RoPE + 9x9 head attention (f32 VALU) ---------
  {
    #pragma unroll 1
    for (int p = tid; p < M_BLOCK * N_HEADS; p += 256) {
      int r = p / N_HEADS;
      int h = p % N_HEADS;

      float qf[HEAD_DIM];
      const __bf16* qrow = qS + r * D_MODEL + h * HEAD_DIM;
      #pragma unroll
      for (int d = 0; d < HEAD_DIM; ++d) qf[d] = (float)qrow[d];

      // RoPE (interleaved pairs) from precomputed LDS table
      const float2* rt = ropeT + r * (HEAD_DIM / 2);
      #pragma unroll
      for (int i = 0; i < HEAD_DIM / 2; ++i) {
        float2 cs = rt[i];
        float a = qf[2 * i], b = qf[2 * i + 1];
        qf[2 * i]     = a * cs.x - b * cs.y;
        qf[2 * i + 1] = a * cs.y + b * cs.x;
      }

      float sc[N_HEADS];
      float mx = -1e30f;
      #pragma unroll
      for (int g = 0; g < N_HEADS; ++g) {
        const __bf16* krow = kS + r * D_MODEL + g * HEAD_DIM;
        float d0 = 0.f;
        #pragma unroll
        for (int d = 0; d < HEAD_DIM; ++d) d0 += qf[d] * (float)krow[d];
        sc[g] = d0 * 0.125f;                 // 1/sqrt(64)
        mx = fmaxf(mx, sc[g]);
      }
      float sum = 0.f;
      #pragma unroll
      for (int g = 0; g < N_HEADS; ++g) { sc[g] = __expf(sc[g] - mx); sum += sc[g]; }
      float inv = 1.0f / sum;

      float acc[HEAD_DIM];
      #pragma unroll
      for (int d = 0; d < HEAD_DIM; ++d) acc[d] = 0.f;
      #pragma unroll
      for (int g = 0; g < N_HEADS; ++g) {
        float w = sc[g] * inv;
        const __bf16* vrow = vS + r * D_MODEL + g * HEAD_DIM;
        #pragma unroll
        for (int d = 0; d < HEAD_DIM; ++d) acc[d] += w * (float)vrow[d];
      }

      // write attn in packed A-fragment layout (aliases xnA; safe post-barrier)
      int mtile = r >> 4, m = r & 15;
      #pragma unroll
      for (int d = 0; d < HEAD_DIM; ++d) {
        int col = h * HEAD_DIM + d;
        int kc  = col >> 5;
        int kk  = col & 31;
        xnA[(mtile * KCH + kc) * FRAG_ELEM + frag_elem_off(m, kk)] = (__bf16)acc[d];
      }
    }
  }
  __syncthreads();

  // ---------------- Stage D: out projection (WMMA) + residual -------------
  {
    #pragma unroll 1
    for (int t = 0; t < 9; ++t) {
      int jb    = wave + 8 * t;   // 0..71
      int mtile = jb & 1;
      int nt    = jb >> 1;        // 0..35
      v8f acc = {};
      const __bf16* wbase = pack + (size_t)(QF + KVF + nt * KCH) * FRAG_ELEM;
      const __bf16* aA    = xnA + mtile * KCH * FRAG_ELEM;
      #pragma unroll
      for (int k = 0; k < KCH; ++k) {
        v16bf a = *(const v16bf*)(aA    + k * FRAG_ELEM + lane * 16);
        v16bf b = *(const v16bf*)(wbase + k * FRAG_ELEM + lane * 16);
        acc = __builtin_amdgcn_wmma_f32_16x16x32_bf16(
            false, a, false, b, (short)0, acc, false, false);
      }
      int col  = nt * 16 + (lane & 15);
      int rowb = mtile * 16 + (lane >> 4) * 8;
      #pragma unroll
      for (int j = 0; j < 8; ++j) {
        size_t gidx = (size_t)(blockTok + rowb + j) * D_MODEL + col;
        out[gidx] = acc[j] + x[gidx];   // residual
      }
    }
  }
}

// ---------------------------------------------------------------------------
extern "C" void kernel_launch(void* const* d_in, const int* in_sizes, int n_in,
                              void* d_out, int out_size, void* d_ws, size_t ws_size,
                              hipStream_t stream) {
  const float* x   = (const float*)d_in[0];
  const float* Wq  = (const float*)d_in[1];
  const float* Wkv = (const float*)d_in[2];
  const float* Wo  = (const float*)d_in[3];
  float* out = (float*)d_out;
  __bf16* pack = (__bf16*)d_ws;   // needs PACK_ELEMS*2 ~ 2.6 MB

  // 1) pack + bf16-convert weights (L2-resident afterwards)
  int packBlocks = (PACK_ELEMS + 255) / 256;
  pack_weights_kernel<<<packBlocks, 256, 0, stream>>>(Wq, Wkv, Wo, pack);

  // 2) fused MLA block: 32768 tokens / 32 per WG = 1024 workgroups
  int nTokens = 16 * SEQ_LEN;
  int grid = nTokens / M_BLOCK;
  mla_fused_kernel<<<grid, 256, LDS_BYTES, stream>>>(x, pack, out);
}
